// VitDetAttention_55327768707836
// MI455X (gfx1250) — compile-verified, matrix operand order checked by reading, not run on hardware
//
#include <hip/hip_runtime.h>
#include <hip/hip_bf16.h>

typedef __attribute__((ext_vector_type(16))) _Float16 v16h;
typedef __attribute__((ext_vector_type(8)))  float    v8f;

union V16H { v16h v; uint4 q[2]; _Float16 h[16]; };

#define TOKENS 50176   // 256*196
#define HID    768
#define QKVN   2304
#define NHEAD  12
#define HD     64
#define SEQ    196
#define SEQT   208     // 13 tiles of 16
#define SEQK   224     // 7 chunks of 32
#define NWAVE  13

#define USE_ASYNC 1

// ---------------------------------------------------------------- helpers
__device__ __forceinline__ unsigned lds_addr32(const void* p) {
    // local->flat addrspace cast keeps the group offset in addr[31:0]
    return (unsigned)(unsigned long long)p;
}
__device__ __forceinline__ void async_b128(unsigned lds, unsigned goff,
                                           unsigned long long base) {
    asm volatile("global_load_async_to_lds_b128 %0, %1, %2"
                 :: "v"(lds), "v"(goff), "s"(base) : "memory");
}
__device__ __forceinline__ void async_wait0() {
    asm volatile("s_wait_asynccnt 0" ::: "memory");
}

// ---------------------------------------------------------------- convert
__global__ void f32_to_f16_kernel(const float* __restrict__ in,
                                  _Float16* __restrict__ out, long n) {
    long i = (long)blockIdx.x * blockDim.x + threadIdx.x;
    long stride = (long)gridDim.x * blockDim.x;
    for (; i < n; i += stride) out[i] = (_Float16)in[i];
}

// ---------------------------------------------------------------- GEMM
// C[M,N] = A[M,K] @ W[N,K]^T + bias.  Block tile 128x128, 8 waves, each wave
// a 32x64 sub-tile (8 WMMA / K-step).  K-step 32, double-buffered LDS filled
// with GLOBAL_LOAD_ASYNC_TO_LDS_B128 (ASYNCcnt) prefetch.
// MODE 0: scatter epilogue into q/k/v f16 (q scaled by 1/8).
// MODE 1: f32 row-major output with bias.
template<int MODE>
__global__ __launch_bounds__(256) void gemm128_kernel(
    const _Float16* __restrict__ A, const _Float16* __restrict__ W,
    const float* __restrict__ bias, float* __restrict__ Cout,
    int K, int N,
    _Float16* __restrict__ qo, _Float16* __restrict__ ko,
    _Float16* __restrict__ vo)
{
    __shared__ _Float16 sbuf[2][2][128 * 32];   // [buf][A|W][row*32+col]

    const int tid = threadIdx.x, wave = tid >> 5, lane = tid & 31;
    const int waveM = wave >> 1, waveN = wave & 1;
    const int m0 = blockIdx.x * 128, n0 = blockIdx.y * 128;
    const int kb = (lane >> 4) * 8;
    const int nk = K / 32;

    // per-thread copy slots: 512 b128 chunks per tile, 2 per thread
    unsigned ldsA[2], ldsW[2], goffA[2], goffW[2];
    #pragma unroll
    for (int j = 0; j < 2; ++j) {
        int c = tid + 256 * j, row = c >> 2, seg = c & 3;
        ldsA[j]  = lds_addr32(&sbuf[0][0][row * 32 + seg * 8]);
        ldsW[j]  = lds_addr32(&sbuf[0][1][row * 32 + seg * 8]);
        goffA[j] = (unsigned)(((size_t)(m0 + row) * K + seg * 8) * 2);
        goffW[j] = (unsigned)(((size_t)(n0 + row) * K + seg * 8) * 2);
    }
    const unsigned long long Abase = (unsigned long long)A;
    const unsigned long long Wbase = (unsigned long long)W;
    const unsigned bufstep = 2 * 128 * 32 * 2;   // bytes between buffers

    auto issue = [&](int kk, int b) {
#if USE_ASYNC
        const unsigned boff = (unsigned)b * bufstep;
        const unsigned koff = (unsigned)kk * 2;
        #pragma unroll
        for (int j = 0; j < 2; ++j) {
            async_b128(ldsA[j] + boff, goffA[j] + koff, Abase);
            async_b128(ldsW[j] + boff, goffW[j] + koff, Wbase);
        }
#else
        #pragma unroll
        for (int j = 0; j < 2; ++j) {
            int c = tid + 256 * j, row = c >> 2, seg = c & 3;
            *(uint4*)(&sbuf[b][0][row * 32 + seg * 8]) =
                *(const uint4*)(A + (size_t)(m0 + row) * K + kk + seg * 8);
            *(uint4*)(&sbuf[b][1][row * 32 + seg * 8]) =
                *(const uint4*)(W + (size_t)(n0 + row) * K + kk + seg * 8);
        }
#endif
    };

    v8f zero = {};
    v8f acc[2][4] = { { zero, zero, zero, zero }, { zero, zero, zero, zero } };

    issue(0, 0);
#if USE_ASYNC
    async_wait0();
#endif
    __syncthreads();

    for (int it = 0; it < nk; ++it) {
        const int cur = it & 1;
        if (it + 1 < nk) issue((it + 1) * 32, cur ^ 1);

        const _Float16* As = sbuf[cur][0];
        const _Float16* Ws = sbuf[cur][1];
        V16H a[2];
        #pragma unroll
        for (int mt = 0; mt < 2; ++mt) {
            const _Float16* p =
                As + (waveM * 32 + mt * 16 + (lane & 15)) * 32;
            a[mt].q[0] = *(const uint4*)(p + kb);
            a[mt].q[1] = *(const uint4*)(p + kb + 16);
        }
        #pragma unroll
        for (int nt = 0; nt < 4; ++nt) {
            V16H b;
            const _Float16* p =
                Ws + (waveN * 64 + nt * 16 + (lane & 15)) * 32;
            b.q[0] = *(const uint4*)(p + kb);
            b.q[1] = *(const uint4*)(p + kb + 16);
            #pragma unroll
            for (int mt = 0; mt < 2; ++mt)
                acc[mt][nt] = __builtin_amdgcn_wmma_f32_16x16x32_f16(
                    false, a[mt].v, false, b.v, (short)0, acc[mt][nt],
                    false, false);
        }
#if USE_ASYNC
        if (it + 1 < nk) async_wait0();
#endif
        __syncthreads();
    }

    const int rloc = (lane >> 4) << 3;
    #pragma unroll
    for (int mt = 0; mt < 2; ++mt) {
        #pragma unroll
        for (int nt = 0; nt < 4; ++nt) {
            const int col = n0 + waveN * 64 + nt * 16 + (lane & 15);
            const float bv = bias[col];
            #pragma unroll
            for (int i = 0; i < 8; ++i) {
                const int row = m0 + waveM * 32 + mt * 16 + rloc + i;
                const float val = acc[mt][nt][i] + bv;
                if (MODE == 0) {
                    int which = col / HID, rem = col % HID;
                    int head = rem >> 6, d = rem & 63;
                    int b_ = row / SEQ, p = row % SEQ;
                    size_t di =
                        ((size_t)(b_ * NHEAD + head) * SEQ + p) * HD + d;
                    if (which == 0)      qo[di] = (_Float16)(val * 0.125f);
                    else if (which == 1) ko[di] = (_Float16)val;
                    else                 vo[di] = (_Float16)val;
                } else {
                    Cout[(size_t)row * N + col] = val;
                }
            }
        }
    }
}

// ---------------------------------------------------------------- attention
// One block per (batch, head). 13 waves x 16 query rows.
__global__ __launch_bounds__(416) void attn_kernel(
    const _Float16* __restrict__ qg, const _Float16* __restrict__ kg,
    const _Float16* __restrict__ vg, const float* __restrict__ rph,
    const float* __restrict__ rpw, _Float16* __restrict__ out)
{
    extern __shared__ _Float16 lds[];
    _Float16* Ksh = lds;                         // [208][64]
    _Float16* Vt  = Ksh + SEQT * HD;             // [64][224] (transposed V)
    _Float16* Psh = Vt + HD * SEQK;              // [13][16][224]
    float*   relh = (float*)(Psh + NWAVE * 16 * SEQK);  // [196][14]
    float*   relw = relh + SEQ * 14;                    // [196][14]

    const int bh = blockIdx.x, tid = threadIdx.x;
    const int wave = tid >> 5, lane = tid & 31;
    const int kb = (lane >> 4) * 8;
    const _Float16* kq = kg + (size_t)bh * SEQ * HD;
    const _Float16* vq = vg + (size_t)bh * SEQ * HD;
    const _Float16* qq = qg + (size_t)bh * SEQ * HD;

    // K tile (zero-padded rows 196..207)
    for (int i = tid; i < (SEQT * HD) / 8; i += 416) {
        int row = i >> 3, c = (i & 7) * 8;
        uint4 val = make_uint4(0, 0, 0, 0);
        if (row < SEQ) val = *(const uint4*)(kq + row * HD + c);
        *(uint4*)(Ksh + row * HD + c) = val;
    }
    // V transposed (zero-padded keys 196..223)
    for (int i = tid; i < HD * SEQK; i += 416) {
        int d = i / SEQK, key = i % SEQK;
        Vt[i] = (key < SEQ) ? vq[key * HD + d] : (_Float16)0.0f;
    }
    // decomposed rel-pos bias (q was pre-scaled by 1/8 -> multiply back)
    for (int idx = tid; idx < 2 * SEQ * 14; idx += 416) {
        int sel = idx >= SEQ * 14;
        int j = sel ? idx - SEQ * 14 : idx;
        int p = j / 14, kc = j % 14;
        int qc = sel ? (p % 14) : (p / 14);
        const float* r = (sel ? rpw : rph) + (qc - kc + 13) * HD;
        const _Float16* qrow = qq + p * HD;
        float s = 0.0f;
        for (int c = 0; c < HD; ++c) s += (float)qrow[c] * r[c];
        (sel ? relw : relh)[p * 14 + kc] = s * 8.0f;
    }

    // Q fragments for this wave's 16 rows
    const int p0 = wave * 16 + (lane & 15);
    const int pcq = p0 < SEQ ? p0 : SEQ - 1;
    const _Float16* qrowp = qq + pcq * HD;
    V16H qa0, qa1;
    qa0.q[0] = *(const uint4*)(qrowp + kb);
    qa0.q[1] = *(const uint4*)(qrowp + kb + 16);
    qa1.q[0] = *(const uint4*)(qrowp + 32 + kb);
    qa1.q[1] = *(const uint4*)(qrowp + 32 + kb + 16);
    __syncthreads();

    // S = (Q*scale) @ K^T : 13 key tiles, K-dim 64 = 2 WMMAs each
    v8f zero = {};
    v8f sc[13];
    #pragma unroll
    for (int kt = 0; kt < 13; ++kt) {
        const int krow = kt * 16 + (lane & 15);
        V16H b0, b1;
        b0.q[0] = *(const uint4*)(Ksh + krow * HD + kb);
        b0.q[1] = *(const uint4*)(Ksh + krow * HD + kb + 16);
        b1.q[0] = *(const uint4*)(Ksh + krow * HD + 32 + kb);
        b1.q[1] = *(const uint4*)(Ksh + krow * HD + 32 + kb + 16);
        v8f z = zero;
        z = __builtin_amdgcn_wmma_f32_16x16x32_f16(false, qa0.v, false, b0.v,
                                                   (short)0, z, false, false);
        z = __builtin_amdgcn_wmma_f32_16x16x32_f16(false, qa1.v, false, b1.v,
                                                   (short)0, z, false, false);
        sc[kt] = z;
    }

    // bias + softmax (row spread over 16 lanes) -> P in LDS (f16)
    const int rloc = (lane >> 4) << 3;
    const int colb = lane & 15;
    for (int i = 0; i < 8; ++i) {
        const int r = rloc + i;
        const int p = wave * 16 + r;
        const int pcl = p < SEQ ? p : 0;
        float sv[13], rowmax = -3.0e38f;
        #pragma unroll
        for (int kt = 0; kt < 13; ++kt) {
            const int key = kt * 16 + colb;
            float x = sc[kt][i];
            if (key < SEQ)
                x += relh[pcl * 14 + key / 14] + relw[pcl * 14 + key % 14];
            else
                x = -3.0e38f;
            sv[kt] = x;
            rowmax = fmaxf(rowmax, x);
        }
        for (int m = 1; m <= 8; m <<= 1)
            rowmax = fmaxf(rowmax, __shfl_xor(rowmax, m, 32));
        float rsum = 0.0f, ev[13];
        #pragma unroll
        for (int kt = 0; kt < 13; ++kt) {
            float e = __expf(sv[kt] - rowmax);
            ev[kt] = e; rsum += e;
        }
        for (int m = 1; m <= 8; m <<= 1) rsum += __shfl_xor(rsum, m, 32);
        const float inv = 1.0f / rsum;
        _Float16* prow = Psh + (wave * 16 + r) * SEQK;
        #pragma unroll
        for (int kt = 0; kt < 13; ++kt)
            prow[kt * 16 + colb] = (_Float16)(ev[kt] * inv);
    }
    { // zero the K-padding columns 208..223
        _Float16* prow = Psh + (wave * 16 + (lane & 15)) * SEQK + SEQT + kb;
        #pragma unroll
        for (int j = 0; j < 8; ++j) prow[j] = (_Float16)0.0f;
    }
    __syncthreads();

    // O = P @ V : 7 K-chunks x 4 output col tiles
    v8f oacc[4] = { zero, zero, zero, zero };
    const _Float16* pwave = Psh + wave * 16 * SEQK;
    #pragma unroll
    for (int kc = 0; kc < 7; ++kc) {
        V16H a;
        const _Float16* pp = pwave + (lane & 15) * SEQK + kc * 32;
        a.q[0] = *(const uint4*)(pp + kb);
        a.q[1] = *(const uint4*)(pp + kb + 16);
        #pragma unroll
        for (int nt = 0; nt < 4; ++nt) {
            V16H b;
            const _Float16* vv = Vt + (nt * 16 + (lane & 15)) * SEQK + kc * 32;
            b.q[0] = *(const uint4*)(vv + kb);
            b.q[1] = *(const uint4*)(vv + kb + 16);
            oacc[nt] = __builtin_amdgcn_wmma_f32_16x16x32_f16(
                false, a.v, false, b.v, (short)0, oacc[nt], false, false);
        }
    }

    // store to (token, 768) f16 layout for the proj GEMM
    const int b_ = bh / NHEAD, head = bh % NHEAD;
    #pragma unroll
    for (int nt = 0; nt < 4; ++nt) {
        #pragma unroll
        for (int i = 0; i < 8; ++i) {
            const int p = wave * 16 + rloc + i;
            if (p < SEQ) {
                const int d = nt * 16 + (lane & 15);
                out[(size_t)(b_ * SEQ + p) * HID + head * HD + d] =
                    (_Float16)oacc[nt][i];
            }
        }
    }
}

// ---------------------------------------------------------------- launch
extern "C" void kernel_launch(void* const* d_in, const int* in_sizes, int n_in,
                              void* d_out, int out_size, void* d_ws, size_t ws_size,
                              hipStream_t stream) {
    const float* hidden  = (const float*)d_in[0];  // (256,14,14,768)
    const float* qkv_w   = (const float*)d_in[1];  // (2304,768)
    const float* qkv_b   = (const float*)d_in[2];  // (2304,)
    const float* proj_w  = (const float*)d_in[3];  // (768,768)
    const float* proj_b  = (const float*)d_in[4];  // (768,)
    const float* rel_h   = (const float*)d_in[5];  // (27,64)
    const float* rel_w   = (const float*)d_in[6];  // (27,64)
    float* outp = (float*)d_out;

    char* ws = (char*)d_ws;
    size_t off = 0;
    _Float16* Wqkv16  = (_Float16*)(ws + off); off += (size_t)QKVN * HID * 2;
    _Float16* Wproj16 = (_Float16*)(ws + off); off += (size_t)HID * HID * 2;
    const size_t qkvsz = (size_t)256 * NHEAD * SEQ * HD * 2;
    _Float16* qf = (_Float16*)(ws + off); off += qkvsz;
    _Float16* kf = (_Float16*)(ws + off); off += qkvsz;
    _Float16* vf = (_Float16*)(ws + off); off += qkvsz;
    // Xf16 (input of GEMM1) and attn16 (output of attention / input of GEMM2)
    // alias the same buffer: GEMM1 finishes reading before attention writes.
    _Float16* Xf16 = (_Float16*)(ws + off); off += (size_t)TOKENS * HID * 2;
    _Float16* attn16 = Xf16;

    // f32 -> f16 conversions
    {
        long nx = (long)TOKENS * HID;
        f32_to_f16_kernel<<<8192, 256, 0, stream>>>(hidden, Xf16, nx);
        long n1 = (long)QKVN * HID;
        f32_to_f16_kernel<<<(unsigned)((n1 + 255) / 256), 256, 0, stream>>>(qkv_w, Wqkv16, n1);
        long n2 = (long)HID * HID;
        f32_to_f16_kernel<<<(unsigned)((n2 + 255) / 256), 256, 0, stream>>>(proj_w, Wproj16, n2);
    }
    // QKV GEMM: (50176x768) @ (2304x768)^T, scatter to q/k/v
    gemm128_kernel<0><<<dim3(TOKENS / 128, QKVN / 128), 256, 0, stream>>>(
        Xf16, Wqkv16, qkv_b, nullptr, HID, QKVN, qf, kf, vf);
    // fused attention per (batch, head)
    {
        size_t shmem = (size_t)(SEQT * HD + HD * SEQK + NWAVE * 16 * SEQK) * 2
                     + (size_t)(2 * SEQ * 14) * 4;
        attn_kernel<<<256 * NHEAD, 416, shmem, stream>>>(qf, kf, vf, rel_h, rel_w, attn16);
    }
    // output projection: (50176x768) @ (768x768)^T -> f32 out
    gemm128_kernel<1><<<dim3(TOKENS / 128, HID / 128), 256, 0, stream>>>(
        attn16, Wproj16, proj_b, outp, HID, HID, nullptr, nullptr, nullptr);
}
